// MaskedReconstructionTask_47364899340912
// MI455X (gfx1250) — compile-verified, matrix operand order checked by reading, not run on hardware
//
#include <hip/hip_runtime.h>
#include <hip/hip_bf16.h>
#include <math.h>

// ---------------------------------------------------------------------------
// MaskedReconstructionTask on MI455X (gfx1250, wave32, WMMA bf16 16x16x32)
//
//   k_init   : zero accumulators
//   k_prep   : pre-swizzle each weight matrix into WMMA B-fragment layout
//   k_stats  : per-feature sum/sumsq + mask count, AND writes masked bf16 x
//              (fuses mask-apply + conversion into the memory-bound pass)
//   k_scale  : inv_scale2[f] = 1/(std_f + 1e-8)^2
//   k_stem   : h1 = GELU(xm @ stem_w + b)                    [WMMA, pure bf16]
//   k_conv   : h2 = GELU(conv1d_k3(h1) + b)                  [WMMA, 3 taps]
//   k_gx     : gx = h2 @ w_ih^T + b_ih, stored (T,B,384)     [WMMA]
//   k_gru    : 4096-step recurrence, batched over B          [WMMA + prefetch]
//   k_head   : fused 3-layer MLP + masked normalized MSE     [WMMA]
//   k_final  : out[0] = loss / max(mask_count, 1)
// ---------------------------------------------------------------------------

typedef __bf16 bf16;
typedef __attribute__((ext_vector_type(16))) __bf16 bf16x16;
typedef __attribute__((ext_vector_type(8)))  __bf16 bf16x8;
typedef __attribute__((ext_vector_type(8)))  float  f32x8;
typedef __attribute__((ext_vector_type(4)))  float  f32x4;
typedef __attribute__((ext_vector_type(8)))  unsigned char u8x8;

#define BB     64
#define TT     4096
#define FF     64
#define DHID   64
#define DG     128
#define ROWS   (BB * TT)     // 262144 flattened (b,t) rows
#define MTILES (ROWS / 16)   // 16384

__device__ __forceinline__ f32x8 wmma_bf16(bf16x16 a, bf16x16 b, f32x8 c) {
  return __builtin_amdgcn_wmma_f32_16x16x32_bf16(
      false, a, false, b, (short)0, c, false, false);
}

// ISA 7.12.2 16-bit fragment K index: lanes 0-15 hold K {0..7,16..23},
// lanes 16-31 hold K {8..15,24..31}.
__device__ __forceinline__ int kmap(int half, int e) {
  return (e < 8 ? e : e + 8) + (half << 3);
}

// Pre-swizzled B-fragment fetch: one 32-byte contiguous load.
__device__ __forceinline__ bf16x16 load_frag(const bf16* __restrict__ base,
                                             int nt, int nks, int ks, int lane) {
  return *(const bf16x16*)(base + ((((size_t)nt * nks + ks) * 32 + lane) << 4));
}

// A-fragment from a row-major bf16 row: two contiguous 16-byte loads.
__device__ __forceinline__ bf16x16 load_arow(const bf16* __restrict__ row,
                                             int k0, int half) {
  bf16x8 lo = *(const bf16x8*)(row + k0 + half * 8);
  bf16x8 hi = *(const bf16x8*)(row + k0 + 16 + half * 8);
  return __builtin_shufflevector(lo, hi, 0, 1, 2, 3, 4, 5, 6, 7,
                                 8, 9, 10, 11, 12, 13, 14, 15);
}

__device__ __forceinline__ float gelu_exact(float v) {
  return 0.5f * v * (1.0f + erff(v * 0.70710678118654752f));
}
__device__ __forceinline__ float sigm(float v) {
  return 1.0f / (1.0f + expf(-v));
}

// ---------------------------------------------------------------------------
__global__ void k_init(float* __restrict__ acc) {
  int i = blockIdx.x * blockDim.x + threadIdx.x;
  if (i < 130) acc[i] = 0.0f;  // f_sum[64], f_sumsq[64], mcount, lossacc
}

// ---------------------------------------------------------------------------
// Pre-swizzle weight matrix into fragment layout. GEMM element (k,n) comes
// from src[k*rs + n*cs]. dst index = ((nt*nks+ks)*32+lane)*16+e.
__global__ void k_prep(const float* __restrict__ src, bf16* __restrict__ dst,
                       int K, int N, int rs, int cs) {
  int idx = blockIdx.x * 256 + threadIdx.x;
  if (idx >= K * N) return;
  int e    = idx & 15;
  int lane = (idx >> 4) & 31;
  int rest = idx >> 9;
  int nks  = K >> 5;
  int ks   = rest % nks;
  int nt   = rest / nks;
  int k = ks * 32 + kmap(lane >> 4, e);
  int n = nt * 16 + (lane & 15);
  dst[idx] = (bf16)src[(size_t)k * rs + (size_t)n * cs];
}

// ---------------------------------------------------------------------------
// Fused: per-feature sum/sumsq (for std), mask count, and masked bf16 copy of
// x. Each thread owns a contiguous 8-feature octet -> all accesses vectorized.
__global__ void __launch_bounds__(256) k_stats(
    const float* __restrict__ x, const unsigned char* __restrict__ msk,
    bf16* __restrict__ xm,
    float* __restrict__ fsum, float* __restrict__ fsumsq, float* __restrict__ mcount) {
  int o   = threadIdx.x & 7;   // feature octet: features [8o, 8o+8)
  int grp = threadIdx.x >> 3;  // 32 row-groups per block
  int f0  = o * 8;
  float s[8] = {}, s2[8] = {};
  float mc = 0.0f;
  for (int g = blockIdx.x * 32 + grp; g < ROWS; g += gridDim.x * 32) {
    const float* p = x + (size_t)g * FF + f0;
    f32x4 v0 = *(const f32x4*)p;
    f32x4 v1 = *(const f32x4*)(p + 4);
    u8x8  mm = *(const u8x8*)(msk + (size_t)g * FF + f0);
    bf16x8 xv;
#pragma unroll
    for (int e = 0; e < 4; ++e) {
      float a = v0[e], b = v1[e];
      s[e] += a;      s2[e] += a * a;
      s[4 + e] += b;  s2[4 + e] += b * b;
      mc += (mm[e] ? 1.0f : 0.0f) + (mm[4 + e] ? 1.0f : 0.0f);
      xv[e]     = mm[e]     ? (bf16)0.0f : (bf16)a;
      xv[4 + e] = mm[4 + e] ? (bf16)0.0f : (bf16)b;
    }
    *(bf16x8*)(xm + (size_t)g * FF + f0) = xv;   // 16-byte masked bf16 store
  }
  __shared__ float sh1[256 * 8], sh2[256 * 8], shm[256];
#pragma unroll
  for (int e = 0; e < 8; ++e) { sh1[threadIdx.x * 8 + e] = s[e]; sh2[threadIdx.x * 8 + e] = s2[e]; }
  shm[threadIdx.x] = mc;
  __syncthreads();
  if (threadIdx.x < 64) {
    int f = threadIdx.x, oo = f >> 3, e = f & 7;
    float a = 0.0f, b = 0.0f;
    for (int gp = 0; gp < 32; ++gp) {
      int t = (gp << 3) | oo;
      a += sh1[t * 8 + e];
      b += sh2[t * 8 + e];
    }
    atomicAdd(&fsum[f], a);
    atomicAdd(&fsumsq[f], b);
  }
  if (threadIdx.x == 0) {
    float t = 0.0f;
    for (int i = 0; i < 256; ++i) t += shm[i];
    atomicAdd(mcount, t);
  }
}

// ---------------------------------------------------------------------------
__global__ void k_scale(const float* __restrict__ fsum, const float* __restrict__ fsumsq,
                        float* __restrict__ inv_s2) {
  int f = threadIdx.x;
  if (f < FF) {
    const float N = (float)ROWS;
    float mean = fsum[f] / N;
    float var  = (fsumsq[f] - N * mean * mean) / (N - 1.0f);  // ddof=1
    float sd   = sqrtf(fmaxf(var, 0.0f)) + 1e-8f;
    inv_s2[f]  = 1.0f / (sd * sd);
  }
}

// ---------------------------------------------------------------------------
// stem: h1 = GELU(xm @ stem_w(64x64) + stem_b) -- pure bf16 WMMA GEMM now.
__global__ void __launch_bounds__(256) k_stem(
    const bf16* __restrict__ xm, const bf16* __restrict__ fragw,
    const float* __restrict__ bias, bf16* __restrict__ out) {
  int lane  = threadIdx.x & 31;
  int wv    = threadIdx.x >> 5;
  int mtile = blockIdx.x * 2 + (wv >> 2);
  int ntile = wv & 3;
  int m = lane & 15, half = lane >> 4;
  int n = ntile * 16 + m;
  int g = mtile * 16 + m;
  const bf16* arow = xm + (size_t)g * FF;
  f32x8 c = {};
#pragma unroll
  for (int ks = 0; ks < 2; ++ks)
    c = wmma_bf16(load_arow(arow, ks * 32, half),
                  load_frag(fragw, ntile, 2, ks, lane), c);
  float bn = bias[n];
#pragma unroll
  for (int r = 0; r < 8; ++r) {
    int gout = mtile * 16 + r + half * 8;
    out[(size_t)gout * DHID + n] = (bf16)gelu_exact(c[r] + bn);
  }
}

// ---------------------------------------------------------------------------
// conv: h2 = GELU(conv1d(h1,k=3,pad=1) + b), 3 shifted WMMA GEMMs.
__global__ void __launch_bounds__(256) k_conv(
    const bf16* __restrict__ hin, const bf16* __restrict__ fragw,
    const float* __restrict__ bias, bf16* __restrict__ out) {
  int lane  = threadIdx.x & 31;
  int wv    = threadIdx.x >> 5;
  int mtile = blockIdx.x * 2 + (wv >> 2);
  int ntile = wv & 3;
  int m = lane & 15, half = lane >> 4;
  int n = ntile * 16 + m;
  int g = mtile * 16 + m;
  int t = g & (TT - 1);
  f32x8 c = {};
#pragma unroll
  for (int d = 0; d < 3; ++d) {
    int ts = t + d - 1;
    bool valid = (unsigned)ts < (unsigned)TT;          // zero pad at batch edges
    const bf16* arow = hin + (size_t)(valid ? (g + d - 1) : g) * DHID;
#pragma unroll
    for (int ks = 0; ks < 2; ++ks) {
      bf16x16 a = load_arow(arow, ks * 32, half);      // unconditional vector loads
#pragma unroll
      for (int e = 0; e < 16; ++e) a[e] = valid ? a[e] : (bf16)0.0f;  // cndmask
      c = wmma_bf16(a, load_frag(fragw + d * (DHID * DHID), ntile, 2, ks, lane), c);
    }
  }
  float bn = bias[n];
#pragma unroll
  for (int r = 0; r < 8; ++r) {
    int gout = mtile * 16 + r + half * 8;
    out[(size_t)gout * DHID + n] = (bf16)gelu_exact(c[r] + bn);
  }
}

// ---------------------------------------------------------------------------
// gx = h2 @ w_ih^T + b_ih, stored transposed as (T, B, 384) so the serial
// GRU loop reads one contiguous 24 KB slab per block-step.
__global__ void __launch_bounds__(256) k_gx(
    const bf16* __restrict__ hin, const bf16* __restrict__ fragw,
    const float* __restrict__ b_ih, float* __restrict__ gx) {
  int lane  = threadIdx.x & 31;
  int wv    = threadIdx.x >> 5;
  int tile  = blockIdx.x * 8 + wv;
  int mtile = tile / 24;
  int ntile = tile % 24;
  int m = lane & 15, half = lane >> 4;
  int n = ntile * 16 + m;
  int g = mtile * 16 + m;
  const bf16* arow = hin + (size_t)g * DHID;
  f32x8 c = {};
#pragma unroll
  for (int ks = 0; ks < 2; ++ks)
    c = wmma_bf16(load_arow(arow, ks * 32, half),
                  load_frag(fragw, ntile, 2, ks, lane), c);
  float bn = b_ih[n];
#pragma unroll
  for (int r = 0; r < 8; ++r) {
    int gout = mtile * 16 + r + half * 8;
    int b = gout >> 12, tt = gout & (TT - 1);
    gx[((size_t)tt * BB + b) * 384 + n] = c[r] + bn;   // (T,B,384)
  }
}

// ---------------------------------------------------------------------------
// GRU recurrence: 4 workgroups x 16 batch rows. Wave w owns hidden columns
// [16w,16w+16) of all three gates; W_hh fragments pinned in VGPRs; hidden
// state in LDS (bf16 copy for A-fragments, f32 copy for the blend).
// gx loads are hoisted ahead of the WMMA block to overlap latency, and each
// thread prefetches the next step's slab into L2 (global_prefetch).
__global__ void __launch_bounds__(256) k_gru(
    const float* __restrict__ gx, const bf16* __restrict__ fraghh,
    const float* __restrict__ b_hh, bf16* __restrict__ zout) {
  __shared__ __align__(16) bf16  Hb[16 * DG];   // 4 KB
  __shared__ __align__(16) float Hf[16 * DG];   // 8 KB
  int lane = threadIdx.x & 31;
  int wv   = threadIdx.x >> 5;
  int m = lane & 15, half = lane >> 4;
  int j  = wv * 16 + m;           // hidden column
  int bs = blockIdx.x * 16;       // batch slice

  // W_hh B-fragments: gate gt -> global ntile gt*8 + wv, nks = 4.
  bf16x16 Bf[3][4];
#pragma unroll
  for (int gt = 0; gt < 3; ++gt)
#pragma unroll
    for (int ks = 0; ks < 4; ++ks)
      Bf[gt][ks] = load_frag(fraghh, gt * 8 + wv, 4, ks, lane);
  float bhr = b_hh[j], bhz = b_hh[DG + j], bhn = b_hh[2 * DG + j];

  for (int i = threadIdx.x; i < 16 * DG; i += 256) { Hb[i] = (bf16)0.0f; Hf[i] = 0.0f; }

  for (int t = 0; t < TT; ++t) {
    __syncthreads();              // H writes from previous step visible
    // Issue this step's gx loads first (overlap with the WMMAs below).
    float gr[8], gz[8], gn[8];
#pragma unroll
    for (int r = 0; r < 8; ++r) {
      int mo = r + half * 8;
      size_t base = ((size_t)t * BB + (bs + mo)) * 384;
      gr[r] = gx[base + j];
      gz[r] = gx[base + DG + j];
      gn[r] = gx[base + 2 * DG + j];
    }
    if (t + 1 < TT)               // prime L2 for next slab (global_prefetch_b8)
      __builtin_prefetch(gx + ((size_t)(t + 1) * BB + bs) * 384 + threadIdx.x * 24, 0, 1);

    f32x8 cr = {}, cz = {}, cn = {};
#pragma unroll
    for (int r = 0; r < 8; ++r) { cr[r] = bhr; cz[r] = bhz; cn[r] = bhn; }
#pragma unroll
    for (int ks = 0; ks < 4; ++ks) {
      bf16x16 a = load_arow(Hb + m * DG, ks * 32, half);  // ds_load_b128 x2
      cr = wmma_bf16(a, Bf[0][ks], cr);
      cz = wmma_bf16(a, Bf[1][ks], cz);
      cn = wmma_bf16(a, Bf[2][ks], cn);
    }
    __syncthreads();              // all Hb reads done before overwrite
#pragma unroll
    for (int r = 0; r < 8; ++r) {
      int mo = r + half * 8;
      float rr = sigm(gr[r] + cr[r]);
      float zz = sigm(gz[r] + cz[r]);
      float nn = tanhf(gn[r] + rr * cn[r]);
      float hold = Hf[mo * DG + j];
      float hnew = (1.0f - zz) * nn + zz * hold;
      Hf[mo * DG + j] = hnew;
      Hb[mo * DG + j] = (bf16)hnew;
      zout[((size_t)(bs + mo) * TT + t) * DG + j] = (bf16)hnew;
    }
  }
}

// ---------------------------------------------------------------------------
// Fused head + masked normalized MSE, one 16-row tile per block.
__global__ void __launch_bounds__(256) k_head(
    const bf16* __restrict__ z,
    const bf16* __restrict__ f1, const float* __restrict__ b1,
    const bf16* __restrict__ f2, const float* __restrict__ b2,
    const bf16* __restrict__ f3, const float* __restrict__ b3,
    const float* __restrict__ x, const unsigned char* __restrict__ msk,
    const float* __restrict__ inv_s2, float* __restrict__ lossacc) {
  __shared__ __align__(16) bf16 r1[16 * 128];
  __shared__ __align__(16) bf16 r2[16 * 128];
  __shared__ float red[256];
  int lane = threadIdx.x & 31;
  int wv   = threadIdx.x >> 5;
  int m = lane & 15, half = lane >> 4;
  int mtile = blockIdx.x;
  int g = mtile * 16 + m;

  { // layer 1: z(16x128) @ h1_w
    int n = wv * 16 + m;
    const bf16* arow = z + (size_t)g * DG;
    f32x8 c = {};
#pragma unroll
    for (int ks = 0; ks < 4; ++ks)
      c = wmma_bf16(load_arow(arow, ks * 32, half), load_frag(f1, wv, 4, ks, lane), c);
    float bn = b1[n];
#pragma unroll
    for (int r = 0; r < 8; ++r)
      r1[(r + half * 8) * 128 + n] = (bf16)gelu_exact(c[r] + bn);
  }
  __syncthreads();
  { // layer 2
    int n = wv * 16 + m;
    f32x8 c = {};
#pragma unroll
    for (int ks = 0; ks < 4; ++ks)
      c = wmma_bf16(load_arow(r1 + m * 128, ks * 32, half), load_frag(f2, wv, 4, ks, lane), c);
    float bn = b2[n];
#pragma unroll
    for (int r = 0; r < 8; ++r)
      r2[(r + half * 8) * 128 + n] = (bf16)gelu_exact(c[r] + bn);
  }
  __syncthreads();
  // layer 3 (N=64 -> waves 0..3; wave-uniform branch keeps EXEC full at WMMA)
  float lsum = 0.0f;
  if (wv < 4) {
    int n = wv * 16 + m;   // feature index
    f32x8 c = {};
#pragma unroll
    for (int ks = 0; ks < 4; ++ks)
      c = wmma_bf16(load_arow(r2 + m * 128, ks * 32, half), load_frag(f3, wv, 4, ks, lane), c);
    float bn  = b3[n];
    float is2 = inv_s2[n];
#pragma unroll
    for (int r = 0; r < 8; ++r) {
      int gout = mtile * 16 + r + half * 8;
      float xr = c[r] + bn;
      float xv = x[(size_t)gout * FF + n];
      float d  = xr - xv;
      float contrib = d * d * is2;
      lsum += msk[(size_t)gout * FF + n] ? contrib : 0.0f;  // select, no branch
    }
  }
  red[threadIdx.x] = lsum;
  __syncthreads();
  for (int s = 128; s > 0; s >>= 1) {
    if (threadIdx.x < s) red[threadIdx.x] += red[threadIdx.x + s];
    __syncthreads();
  }
  if (threadIdx.x == 0) atomicAdd(lossacc, red[0]);
}

// ---------------------------------------------------------------------------
__global__ void k_final(const float* __restrict__ lossacc,
                        const float* __restrict__ mcount, float* __restrict__ out) {
  if (threadIdx.x == 0) out[0] = lossacc[0] / fmaxf(mcount[0], 1.0f);
}

// ---------------------------------------------------------------------------
extern "C" void kernel_launch(void* const* d_in, const int* in_sizes, int n_in,
                              void* d_out, int out_size, void* d_ws, size_t ws_size,
                              hipStream_t stream) {
  const float*         x     = (const float*)d_in[0];
  const unsigned char* fmask = (const unsigned char*)d_in[1];
  const float* stem_w = (const float*)d_in[2];
  const float* stem_b = (const float*)d_in[3];
  const float* conv_w = (const float*)d_in[4];
  const float* conv_b = (const float*)d_in[5];
  const float* w_ih   = (const float*)d_in[6];
  const float* w_hh   = (const float*)d_in[7];
  const float* b_ih   = (const float*)d_in[8];
  const float* b_hh   = (const float*)d_in[9];
  const float* h1_w   = (const float*)d_in[10];
  const float* h1_b   = (const float*)d_in[11];
  const float* h2_w   = (const float*)d_in[12];
  const float* h2_b   = (const float*)d_in[13];
  const float* h3_w   = (const float*)d_in[14];
  const float* h3_b   = (const float*)d_in[15];

  // ws layout (floats): [0,64) f_sum, [64,128) f_sumsq, 128 mcount,
  // 129 lossacc, [130,194) inv_scale2; big buffers from +256 floats.
  float* ws      = (float*)d_ws;
  float* f_sum   = ws;
  float* f_sumsq = ws + 64;
  float* mcount  = ws + 128;
  float* lossacc = ws + 129;
  float* inv_s2  = ws + 130;
  bf16*  h1buf   = (bf16*)(ws + 256);                       // 32 MB
  bf16*  h2buf   = h1buf + (size_t)ROWS * DHID;             // 32 MB
  float* gxbuf   = (float*)(h2buf + (size_t)ROWS * DHID);   // 402 MB (T,B,384)
  bf16*  zbuf    = (bf16*)(gxbuf + (size_t)ROWS * 384);     // 64 MB
  bf16*  xmbuf   = zbuf + (size_t)ROWS * DG;                // 32 MB masked bf16 x
  // pre-swizzled weight fragments (bf16), all sizes multiples of 4096 -> aligned
  bf16* frag_stem = xmbuf + (size_t)ROWS * FF;              // 64*64
  bf16* frag_conv = frag_stem + 64 * 64;                    // 3 * 64*64
  bf16* frag_ih   = frag_conv + 3 * 64 * 64;                // 64*384
  bf16* frag_hh   = frag_ih + 64 * 384;                     // 128*384
  bf16* frag_h1   = frag_hh + 128 * 384;                    // 128*128
  bf16* frag_h2   = frag_h1 + 128 * 128;                    // 128*128
  bf16* frag_h3   = frag_h2 + 128 * 128;                    // 128*64

  hipLaunchKernelGGL(k_init, dim3(1), dim3(256), 0, stream, ws);
  // GEMM element (k,n) = src[k*rs + n*cs]
  hipLaunchKernelGGL(k_prep, dim3(16),  dim3(256), 0, stream, stem_w,    frag_stem, 64, 64, 64, 1);
  hipLaunchKernelGGL(k_prep, dim3(16),  dim3(256), 0, stream, conv_w+0,  frag_conv + 0*4096, 64, 64, 3, 192);
  hipLaunchKernelGGL(k_prep, dim3(16),  dim3(256), 0, stream, conv_w+1,  frag_conv + 1*4096, 64, 64, 3, 192);
  hipLaunchKernelGGL(k_prep, dim3(16),  dim3(256), 0, stream, conv_w+2,  frag_conv + 2*4096, 64, 64, 3, 192);
  hipLaunchKernelGGL(k_prep, dim3(96),  dim3(256), 0, stream, w_ih,      frag_ih,   64, 384, 1, 64);
  hipLaunchKernelGGL(k_prep, dim3(192), dim3(256), 0, stream, w_hh,      frag_hh,   128, 384, 1, 128);
  hipLaunchKernelGGL(k_prep, dim3(64),  dim3(256), 0, stream, h1_w,      frag_h1,   128, 128, 128, 1);
  hipLaunchKernelGGL(k_prep, dim3(64),  dim3(256), 0, stream, h2_w,      frag_h2,   128, 128, 128, 1);
  hipLaunchKernelGGL(k_prep, dim3(32),  dim3(256), 0, stream, h3_w,      frag_h3,   128, 64, 64, 1);

  hipLaunchKernelGGL(k_stats, dim3(1024),       dim3(256), 0, stream,
                     x, fmask, xmbuf, f_sum, f_sumsq, mcount);
  hipLaunchKernelGGL(k_scale, dim3(1),          dim3(64),  0, stream,
                     f_sum, f_sumsq, inv_s2);
  hipLaunchKernelGGL(k_stem,  dim3(MTILES / 2), dim3(256), 0, stream,
                     xmbuf, frag_stem, stem_b, h1buf);
  hipLaunchKernelGGL(k_conv,  dim3(MTILES / 2), dim3(256), 0, stream,
                     h1buf, frag_conv, conv_b, h2buf);
  hipLaunchKernelGGL(k_gx,    dim3(MTILES * 3), dim3(256), 0, stream,
                     h2buf, frag_ih, b_ih, gxbuf);
  hipLaunchKernelGGL(k_gru,   dim3(4),          dim3(256), 0, stream,
                     gxbuf, frag_hh, b_hh, zbuf);
  hipLaunchKernelGGL(k_head,  dim3(MTILES),     dim3(256), 0, stream,
                     zbuf, frag_h1, h1_b, frag_h2, h2_b, frag_h3, h3_b,
                     x, fmask, inv_s2, lossacc);
  hipLaunchKernelGGL(k_final, dim3(1),          dim3(1),   0, stream,
                     lossacc, mcount, (float*)d_out);
}
